// APPNPModel_78795470012807
// MI455X (gfx1250) — matrix operand order repaired and checked by reading, not compiled, not run on hardware
//
#include <hip/hip_runtime.h>
#include <hip/hip_bf16.h>
#include <stdint.h>

// ---------------------------------------------------------------------------
// APPNP on MI455X (gfx1250, wave32, WMMA)
//   h0 = relu(x@W1+b1)@W2+b2   -> fused bf16 WMMA kernel; W1 staged into LDS
//                                 via global_load_async_to_lds_b128 (ASYNCcnt,
//                                 double-buffered), 32 rows/wave.
//   10x propagation            -> L2-resident gather + global_atomic_add_f32
//   softmax rows of 64         -> wave32 shuffle reductions
// ---------------------------------------------------------------------------

typedef __bf16 bf16_t;
typedef __attribute__((ext_vector_type(16))) __bf16 v16bf;
typedef __attribute__((ext_vector_type(8)))  __bf16 v8bf;
typedef __attribute__((ext_vector_type(8)))  float  v8f;
typedef __attribute__((ext_vector_type(4)))  int    v4i;

#define N_FEAT   500
#define N_FEAT_P 512
#define HIDDEN   128
#define N_CLS    64
#define K_LAYERS 10
#define ALPHA_F  0.1f

#define LDSW1_STRIDE 40    // 32 data + 8 pad bf16 -> 20 dwords: conflict-free reads
#define LDSH_STRIDE  136   // 128 data + 8 pad bf16 -> 68 dwords: conflict-free reads

#if defined(__HIP_DEVICE_COMPILE__) && __has_builtin(__builtin_amdgcn_global_load_async_to_lds_b128)
#define HAVE_ASYNC_LDS 1
#else
#define HAVE_ASYNC_LDS 0
#endif

__device__ inline v16bf cat16(v8bf lo, v8bf hi) {
  return __builtin_shufflevector(lo, hi, 0,1,2,3,4,5,6,7,8,9,10,11,12,13,14,15);
}

#if HAVE_ASYNC_LDS
__device__ inline void async_copy_b128(const bf16_t* g, bf16_t* l) {
  // builtin signature (from hipcc diagnostic): (v4i*, v4i*, imm offset, imm cpol)
  __builtin_amdgcn_global_load_async_to_lds_b128((v4i*)g, (v4i*)l, 0, 0);
}
__device__ inline void wait_async0() {
#if __has_builtin(__builtin_amdgcn_s_wait_asynccnt)
  __builtin_amdgcn_s_wait_asynccnt(0);
#else
  asm volatile("s_wait_asynccnt 0x0" ::: "memory");
#endif
}
#endif

// ---------------------------------------------------------------------------
// Packing kernels: bf16 convert + pad + transpose weights
// ---------------------------------------------------------------------------
__global__ __launch_bounds__(256) void pack_x_kernel(const float* __restrict__ x,
                                                     bf16_t* __restrict__ xbf,
                                                     int nnodes) {
  long long tid = (long long)blockIdx.x * 256 + threadIdx.x;
  long long total = (long long)nnodes * N_FEAT_P;
  if (tid >= total) return;
  int n = (int)(tid >> 9);         // / 512
  int k = (int)(tid & (N_FEAT_P - 1));
  float v = (k < N_FEAT) ? x[(long long)n * N_FEAT + k] : 0.0f;
  xbf[tid] = (bf16_t)v;
}

// w1t: [HIDDEN][512] = W1^T zero-padded ; w2t: [N_CLS][HIDDEN] = W2^T
__global__ __launch_bounds__(256) void pack_w_kernel(const float* __restrict__ W1,
                                                     const float* __restrict__ W2,
                                                     bf16_t* __restrict__ w1t,
                                                     bf16_t* __restrict__ w2t) {
  int tid = blockIdx.x * 256 + threadIdx.x;
  const int W1T_ELEMS = HIDDEN * N_FEAT_P;          // 65536
  const int W2T_ELEMS = N_CLS * HIDDEN;             // 8192
  if (tid < W1T_ELEMS) {
    int n = tid >> 9;                               // hidden col
    int k = tid & (N_FEAT_P - 1);
    float v = (k < N_FEAT) ? W1[(long long)k * HIDDEN + n] : 0.0f;
    w1t[tid] = (bf16_t)v;
  } else if (tid < W1T_ELEMS + W2T_ELEMS) {
    int u = tid - W1T_ELEMS;
    int n = u >> 7;                                 // class col
    int k = u & (HIDDEN - 1);
    w2t[u] = (bf16_t)W2[(long long)k * N_CLS + n];
  }
}

// ---------------------------------------------------------------------------
// Fused MLP: h0[n][64] = relu(x@W1+b1)@W2+b2.
// Block = 256 threads = 8 waves; each wave owns 32 rows (two 16-row A tiles),
// so every B fragment pulled from LDS feeds TWO WMMAs. W1 staging is double-
// buffered async DMA (global_load_async_to_lds_b128): chunk k+1's DMA runs
// during chunk k's WMMAs, waited with s_wait_asynccnt at the next iteration.
// No early wave-return (barriers inside the K loop).
//
// A fragment (16x32 bf16): lane (m = lane&15, h = lane>>4),
//   elems 0..7  <- K = kb + 8h + 0..7, elems 8..15 <- K = kb + 16 + 8h + 0..7
// B fragment (32x16 bf16): lane (n = lane&15, h = lane>>4),
//   elems 0..7 <- K = kb + 8h + 0..7, elems 8..15 <- K = kb + 16 + 8h + 0..7 (same as A)
// C/D: elem m -> row m + 8h, col = lane&15 (per 16x16 tile)
// ---------------------------------------------------------------------------
__global__ __launch_bounds__(256) void mlp_kernel(const bf16_t* __restrict__ xbf,
                                                  const bf16_t* __restrict__ w1t,
                                                  const float*  __restrict__ b1,
                                                  const bf16_t* __restrict__ w2t,
                                                  const float*  __restrict__ b2,
                                                  float* __restrict__ h0,
                                                  int nnodes) {
  __shared__ bf16_t sW1[2][HIDDEN * LDSW1_STRIDE];      // 2 x 10.0 KB W1 chunks
  __shared__ bf16_t sH[8 * 16 * LDSH_STRIDE];           // 34.0 KB hidden tiles

  const int wave = threadIdx.x >> 5;
  const int lane = threadIdx.x & 31;
  const int nl   = lane & 15;
  const int h    = lane >> 4;
  const int row0 = (blockIdx.x * 8 + wave) << 5;        // 32 rows per wave
  int r0 = row0 + nl;      if (r0 >= nnodes) r0 = nnodes - 1;
  int r1 = row0 + 16 + nl; if (r1 >= nnodes) r1 = nnodes - 1;
  const bf16_t* arow0 = xbf + (long long)r0 * N_FEAT_P;
  const bf16_t* arow1 = xbf + (long long)r1 * N_FEAT_P;

  // ---- GEMM1: two 16x128 tiles, K = 512, pipelined W1 staging ----
  v8f c0[8] = {};
  v8f c1[8] = {};
  const int sn     = threadIdx.x >> 1;                  // 0..127 (W1 column)
  const int shalf  = threadIdx.x & 1;
  const int ldsIdx = sn * LDSW1_STRIDE + shalf * 16;
  const bf16_t* sp_base = w1t + (long long)sn * N_FEAT_P + shalf * 16;

#if HAVE_ASYNC_LDS
  // prologue: DMA chunk 0 into buffer 0 (two b128 per thread = 32 bytes)
  async_copy_b128(sp_base,     &sW1[0][ldsIdx]);
  async_copy_b128(sp_base + 8, &sW1[0][ldsIdx] + 8);
#else
  v8bf st0 = *(const v8bf*)(sp_base);                   // prologue: chunk 0
  v8bf st1 = *(const v8bf*)(sp_base + 8);
#endif

  for (int kb = 0; kb < N_FEAT_P; kb += 32) {
#if HAVE_ASYNC_LDS
    const int cur = (kb >> 5) & 1;
    wait_async0();                                      // my DMA of chunk kb done
    __syncthreads();                                    // everyone's DMA done
    if (kb + 32 < N_FEAT_P) {                           // DMA chunk kb+32 -> other buf
      bf16_t* nb = &sW1[cur ^ 1][ldsIdx];
      async_copy_b128(sp_base + kb + 32, nb);
      async_copy_b128(sp_base + kb + 40, nb + 8);
    }
    const bf16_t* bROW = &sW1[cur][0];
#else
    __syncthreads();                                    // prev readers done
    bf16_t* dp = &sW1[0][ldsIdx];
    *(v8bf*)dp       = st0;
    *(v8bf*)(dp + 8) = st1;
    __syncthreads();                                    // chunk kb visible
    if (kb + 32 < N_FEAT_P) {                           // prefetch chunk kb+32
      st0 = *(const v8bf*)(sp_base + kb + 32);
      st1 = *(const v8bf*)(sp_base + kb + 40);
    }
    const bf16_t* bROW = &sW1[0][0];
#endif
    v16bf a0 = cat16(*(const v8bf*)(arow0 + kb + 8 * h),
                     *(const v8bf*)(arow0 + kb + 16 + 8 * h));
    v16bf a1 = cat16(*(const v8bf*)(arow1 + kb + 8 * h),
                     *(const v8bf*)(arow1 + kb + 16 + 8 * h));
    // B fragments with one-tile lookahead: DS returns in order, so each WMMA
    // pair waits dscnt<=2 instead of 0.
    const bf16_t* bbase = bROW + nl * LDSW1_STRIDE;
    v8bf p0 = *(const v8bf*)(bbase + 8 * h);
    v8bf p1 = *(const v8bf*)(bbase + 16 + 8 * h);
#pragma unroll
    for (int t = 0; t < 8; t++) {
      v16bf bfrag = cat16(p0, p1);
      if (t < 7) {
        const bf16_t* np = bbase + (t + 1) * 16 * LDSW1_STRIDE;
        p0 = *(const v8bf*)(np + 8 * h);
        p1 = *(const v8bf*)(np + 16 + 8 * h);
      }
      c0[t] = __builtin_amdgcn_wmma_f32_16x16x32_bf16(false, a0, false, bfrag,
                                                      (short)0, c0[t], false, false);
      c1[t] = __builtin_amdgcn_wmma_f32_16x16x32_bf16(false, a1, false, bfrag,
                                                      (short)0, c1[t], false, false);
    }
  }

  // ---- two passes: (bias+ReLU -> wave-private LDS -> GEMM2 -> store h0) ----
  // Wave-private LDS region; LDS ops are in-order within a wave (DScnt).
  bf16_t* myH = sH + wave * 16 * LDSH_STRIDE;
  const bf16_t* harow = myH + nl * LDSH_STRIDE;
#pragma unroll
  for (int ht = 0; ht < 2; ht++) {
    // hidden tile (C-layout scatter into LDS, bf16)
#pragma unroll
    for (int t = 0; t < 8; t++) {
      float bias = b1[t * 16 + nl];
#pragma unroll
      for (int m = 0; m < 8; m++) {
        float v = (ht ? c1[t][m] : c0[t][m]) + bias;
        v = v > 0.0f ? v : 0.0f;
        myH[(m + 8 * h) * LDSH_STRIDE + t * 16 + nl] = (bf16_t)v;
      }
    }
    // GEMM2: 16x64 tile, K = 128, A fragments re-read from LDS
    v8f c2[4] = {};
#pragma unroll
    for (int kb = 0; kb < HIDDEN; kb += 32) {
      v16bf a = cat16(*(const v8bf*)(harow + kb + 8 * h),
                      *(const v8bf*)(harow + kb + 16 + 8 * h));
#pragma unroll
      for (int t = 0; t < 4; t++) {
        const bf16_t* bp = w2t + (long long)(t * 16 + nl) * HIDDEN + kb + 8 * h;
        v16bf bfrag = cat16(*(const v8bf*)bp, *(const v8bf*)(bp + 16));
        c2[t] = __builtin_amdgcn_wmma_f32_16x16x32_bf16(false, a, false, bfrag,
                                                        (short)0, c2[t], false, false);
      }
    }
#pragma unroll
    for (int t = 0; t < 4; t++) {
      float bias = b2[t * 16 + nl];
#pragma unroll
      for (int m = 0; m < 8; m++) {
        int row = row0 + ht * 16 + m + 8 * h;
        if (row < nnodes)
          h0[(long long)row * N_CLS + t * 16 + nl] = c2[t][m] + bias;
      }
    }
  }
}

// ---------------------------------------------------------------------------
// Degree / normalization
// ---------------------------------------------------------------------------
__global__ __launch_bounds__(256) void deg_init_kernel(float* __restrict__ deg, int n) {
  int i = blockIdx.x * 256 + threadIdx.x;
  if (i < n) deg[i] = 1.0f;                         // self-loop contribution
}

__global__ __launch_bounds__(256) void edge_cvt_kernel(const long long* __restrict__ ei,
                                                       int* __restrict__ src32,
                                                       int* __restrict__ dst32,
                                                       float* __restrict__ deg,
                                                       int nedges) {
  int e = blockIdx.x * 256 + threadIdx.x;
  if (e >= nedges) return;
  int s = (int)ei[e];
  int d = (int)ei[(long long)nedges + e];
  src32[e] = s;
  dst32[e] = d;
  unsafeAtomicAdd(&deg[d], 1.0f);                   // hardware global_atomic_add_f32
}

__global__ __launch_bounds__(256) void dinv_kernel(const float* __restrict__ deg,
                                                   float* __restrict__ dinv, int n) {
  int i = blockIdx.x * 256 + threadIdx.x;
  if (i < n) {
    float dg = deg[i];
    dinv[i] = dg > 0.0f ? rsqrtf(dg) : 0.0f;
  }
}

// ---------------------------------------------------------------------------
// Propagation step = init pass (alpha*h0 + self-loop term) + edge atomics
// ---------------------------------------------------------------------------
__global__ __launch_bounds__(256) void prop_init_kernel(const float* __restrict__ h0,
                                                        const float* __restrict__ zin,
                                                        const float* __restrict__ dinv,
                                                        float* __restrict__ zout,
                                                        int nnodes) {
  long long tid = (long long)blockIdx.x * 256 + threadIdx.x;
  if (tid >= (long long)nnodes * N_CLS) return;
  int n = (int)(tid >> 6);
  float di = dinv[n];
  zout[tid] = ALPHA_F * h0[tid] + (1.0f - ALPHA_F) * di * di * zin[tid];
}

// one wave per edge; lane handles 2 channels (float2 gather + 2 f32 atomics).
// edge id is wave-uniform -> scalarize src/dst/weight via readfirstlane.
__global__ __launch_bounds__(256) void prop_edge_kernel(const int* __restrict__ src32,
                                                        const int* __restrict__ dst32,
                                                        const float* __restrict__ dinv,
                                                        const float* __restrict__ zin,
                                                        float* __restrict__ zout,
                                                        int nedges) {
  long long tid = (long long)blockIdx.x * 256 + threadIdx.x;
  int e = (int)(tid >> 5);                          // wave-uniform
  if (e >= nedges) return;
  int l = (int)(tid & 31);
  if (e + 4096 < nedges) {                          // stream-ahead: global_prefetch_b8
    __builtin_prefetch(src32 + e + 4096, 0, 1);
    __builtin_prefetch(dst32 + e + 4096, 0, 1);
  }
  int s = __builtin_amdgcn_readfirstlane(src32[e]);
  int d = __builtin_amdgcn_readfirstlane(dst32[e]);
  float w = (1.0f - ALPHA_F) * dinv[s] * dinv[d];
  const float2 v = *(const float2*)(zin + (long long)s * N_CLS + l * 2);
  float* op = zout + (long long)d * N_CLS + l * 2;
  unsafeAtomicAdd(op,     w * v.x);
  unsafeAtomicAdd(op + 1, w * v.y);
}

// ---------------------------------------------------------------------------
// Row softmax over 64 classes: one wave per node
// ---------------------------------------------------------------------------
__global__ __launch_bounds__(256) void softmax_kernel(const float* __restrict__ z,
                                                      float* __restrict__ out,
                                                      int nnodes) {
  int wave = threadIdx.x >> 5;
  int lane = threadIdx.x & 31;
  int n = blockIdx.x * 8 + wave;
  if (n >= nnodes) return;
  const float2 v = *(const float2*)(z + (long long)n * N_CLS + lane * 2);
  float mx = fmaxf(v.x, v.y);
#pragma unroll
  for (int m = 16; m >= 1; m >>= 1) mx = fmaxf(mx, __shfl_xor(mx, m, 32));
  float e0 = __expf(v.x - mx);
  float e1 = __expf(v.y - mx);
  float s = e0 + e1;
#pragma unroll
  for (int m = 16; m >= 1; m >>= 1) s += __shfl_xor(s, m, 32);
  float inv = 1.0f / s;
  float* op = out + (long long)n * N_CLS + lane * 2;
  op[0] = e0 * inv;
  op[1] = e1 * inv;
}

// ---------------------------------------------------------------------------
// Host launcher
// ---------------------------------------------------------------------------
extern "C" void kernel_launch(void* const* d_in, const int* in_sizes, int n_in,
                              void* d_out, int out_size, void* d_ws, size_t ws_size,
                              hipStream_t stream) {
  const float*     x  = (const float*)d_in[0];
  const long long* ei = (const long long*)d_in[1];   // int64 edge_index [2][E]
  const float*     W1 = (const float*)d_in[2];
  const float*     b1 = (const float*)d_in[3];
  const float*     W2 = (const float*)d_in[4];
  const float*     b2 = (const float*)d_in[5];
  float* out = (float*)d_out;

  const int N = in_sizes[0] / N_FEAT;                // 100000
  const int E = in_sizes[1] / 2;                     // 3200000

  // workspace carve-up (~206 MB total)
  char* p = (char*)d_ws;
  auto take = [&](size_t bytes) -> char* {
    char* r = p;
    p += (bytes + 255) & ~(size_t)255;
    return r;
  };
  bf16_t* xbf   = (bf16_t*)take((size_t)N * N_FEAT_P * sizeof(bf16_t));
  bf16_t* w1t   = (bf16_t*)take((size_t)HIDDEN * N_FEAT_P * sizeof(bf16_t));
  bf16_t* w2t   = (bf16_t*)take((size_t)N_CLS * HIDDEN * sizeof(bf16_t));
  float*  h0    = (float*)take((size_t)N * N_CLS * sizeof(float));
  float*  zA    = (float*)take((size_t)N * N_CLS * sizeof(float));
  float*  zB    = (float*)take((size_t)N * N_CLS * sizeof(float));
  float*  deg   = (float*)take((size_t)N * sizeof(float));
  float*  dinv  = (float*)take((size_t)N * sizeof(float));
  int*    src32 = (int*)take((size_t)E * sizeof(int));
  int*    dst32 = (int*)take((size_t)E * sizeof(int));

  const int B = 256;
  // 1. pack x (bf16, K padded to 512)
  {
    long long tot = (long long)N * N_FEAT_P;
    pack_x_kernel<<<(unsigned)((tot + B - 1) / B), B, 0, stream>>>(x, xbf, N);
  }
  // 2. pack weights (transposed bf16)
  {
    int tot = HIDDEN * N_FEAT_P + N_CLS * HIDDEN;
    pack_w_kernel<<<(tot + B - 1) / B, B, 0, stream>>>(W1, W2, w1t, w2t);
  }
  // 3. degree + int32 edges + dinv
  deg_init_kernel<<<(N + B - 1) / B, B, 0, stream>>>(deg, N);
  edge_cvt_kernel<<<(E + B - 1) / B, B, 0, stream>>>(ei, src32, dst32, deg, E);
  dinv_kernel<<<(N + B - 1) / B, B, 0, stream>>>(deg, dinv, N);
  // 4. fused MLP via WMMA (256 rows per 256-thread block)
  mlp_kernel<<<(N + 255) / 256, B, 0, stream>>>(xbf, w1t, b1, w2t, b2, h0, N);
  // 5. K propagation steps (ping-pong zA/zB, self-loops folded into init)
  const float* zfinal = h0;
  for (int i = 0; i < K_LAYERS; i++) {
    const float* zin  = (i == 0) ? h0 : ((i & 1) ? zA : zB);
    float*       zout = (i & 1) ? zB : zA;
    {
      long long tot = (long long)N * N_CLS;
      prop_init_kernel<<<(unsigned)((tot + B - 1) / B), B, 0, stream>>>(h0, zin, dinv, zout, N);
    }
    {
      long long tot = (long long)E * 32;
      prop_edge_kernel<<<(unsigned)((tot + B - 1) / B), B, 0, stream>>>(src32, dst32, dinv,
                                                                        zin, zout, E);
    }
    zfinal = zout;
  }
  // 6. softmax -> d_out
  softmax_kernel<<<(N + 7) / 8, B, 0, stream>>>(zfinal, out, N);
}
